// TransformerBlock_30056181138107
// MI455X (gfx1250) — matrix operand order repaired and checked by reading, not compiled, not run on hardware
//
#include <hip/hip_runtime.h>
#include <hip/hip_bf16.h>
#include <math.h>

typedef __bf16 bf16_t;
typedef __attribute__((ext_vector_type(16))) __bf16 v16bf;
typedef __attribute__((ext_vector_type(8)))  __bf16 v8bf;
typedef __attribute__((ext_vector_type(8)))  float  v8f;

#if __has_builtin(__builtin_amdgcn_tanhf)
static __device__ __forceinline__ float fast_tanh(float x) { return __builtin_amdgcn_tanhf(x); }
#else
static __device__ __forceinline__ float fast_tanh(float x) { return tanhf(x); }
#endif

// ---------------------------------------------------------------------------
// WMMA fragment helpers (CDNA5 wave32, 16x16x32 bf16, f32 accumulate)
// A (16x32 bf16): lane h=lane>>4, m=lane&15 holds row m, K = h*16 .. h*16+15 (contiguous)
// B (32x16 bf16): lane (h, n) holds column n, K = h*16 .. h*16+15 (contiguous)
// C/D (16x16 f32): reg r holds row r+8h, column = lane&15
// ---------------------------------------------------------------------------
static __device__ __forceinline__ v16bf load_Afrag(const bf16_t* A, int lda) {
  const int lane = threadIdx.x & 31;
  const int h = lane >> 4, m = lane & 15;
  const bf16_t* p = A + (size_t)m * lda + h * 16;
  v8bf lo = *(const v8bf*)p;          // K = h*16 .. h*16+7
  v8bf hi = *(const v8bf*)(p + 8);    // K = h*16+8 .. h*16+15
  v16bf a;
#pragma unroll
  for (int i = 0; i < 8; ++i) { a[i] = lo[i]; a[i + 8] = hi[i]; }
  return a;
}

static __device__ __forceinline__ v16bf load_Bfrag(const bf16_t* W, int ldw) {
  const int lane = threadIdx.x & 31;
  const int h = lane >> 4, n = lane & 15;
  return *(const v16bf*)(W + (size_t)n * ldw + h * 16);  // 32 contiguous bytes
}

static __device__ __forceinline__ v8f wmma_bf16(v16bf a, v16bf b, v8f c) {
  return __builtin_amdgcn_wmma_f32_16x16x32_bf16(false, a, false, b, (short)0, c,
                                                 false, false);
}

// 32x64 output block per wave: 2 A-frags + 4 B-frags feed 8 WMMAs per K-step
// (1.5 b128 loads per WMMA instead of 4 -> VMEM-issue no longer the bottleneck)
static __device__ __forceinline__ void gemm_block_2x4(const bf16_t* A, int lda,
                                                      const bf16_t* W, int ldw,
                                                      int K, v8f acc[2][4]) {
  for (int k0 = 0; k0 < K; k0 += 32) {
    v16bf a0 = load_Afrag(A + k0, lda);
    v16bf a1 = load_Afrag(A + (size_t)16 * lda + k0, lda);
    v16bf b0 = load_Bfrag(W + k0, ldw);
    v16bf b1 = load_Bfrag(W + (size_t)16 * ldw + k0, ldw);
    v16bf b2 = load_Bfrag(W + (size_t)32 * ldw + k0, ldw);
    v16bf b3 = load_Bfrag(W + (size_t)48 * ldw + k0, ldw);
    acc[0][0] = wmma_bf16(a0, b0, acc[0][0]);
    acc[0][1] = wmma_bf16(a0, b1, acc[0][1]);
    acc[0][2] = wmma_bf16(a0, b2, acc[0][2]);
    acc[0][3] = wmma_bf16(a0, b3, acc[0][3]);
    acc[1][0] = wmma_bf16(a1, b0, acc[1][0]);
    acc[1][1] = wmma_bf16(a1, b1, acc[1][1]);
    acc[1][2] = wmma_bf16(a1, b2, acc[1][2]);
    acc[1][3] = wmma_bf16(a1, b3, acc[1][3]);
  }
}

#define ACC_INIT(acc)                                   \
  v8f acc[2][4];                                        \
  _Pragma("unroll") for (int i = 0; i < 2; ++i)         \
  _Pragma("unroll") for (int j = 0; j < 4; ++j) acc[i][j] = (v8f){};

// ---------------------------------------------------------------------------
// Elementwise conversion / transpose kernels (one-time per launch, tiny cost)
// ---------------------------------------------------------------------------
__global__ void k_f32_to_bf16(const float* __restrict__ in, bf16_t* __restrict__ out,
                              int n) {
  int i = blockIdx.x * blockDim.x + threadIdx.x;
  if (i < n) out[i] = (bf16_t)in[i];
}

// out[c*R + r] = in[r*C + c]   (produce (N,K)-major bf16 weights)
__global__ void k_transpose_bf16(const float* __restrict__ in, bf16_t* __restrict__ out,
                                 int R, int C) {
  int c = blockIdx.x * 16 + threadIdx.x;
  int r = blockIdx.y * 16 + threadIdx.y;
  if (r < R && c < C) out[(size_t)c * R + r] = (bf16_t)in[(size_t)r * C + c];
}

// ---------------------------------------------------------------------------
// LayerNorm over D=768, one row per 256-thread block, bf16 output
// ---------------------------------------------------------------------------
__global__ void k_layernorm(const float* __restrict__ X, const float* __restrict__ gamma,
                            const float* __restrict__ beta, bf16_t* __restrict__ out) {
  const int row = blockIdx.x;
  const float* x = X + (size_t)row * 768;
  __shared__ float s1[256], s2[256];
  const int t = threadIdx.x;
  float v0 = x[t], v1 = x[t + 256], v2 = x[t + 512];
  s1[t] = v0 + v1 + v2;
  s2[t] = v0 * v0 + v1 * v1 + v2 * v2;
  __syncthreads();
  for (int s = 128; s > 0; s >>= 1) {
    if (t < s) { s1[t] += s1[t + s]; s2[t] += s2[t + s]; }
    __syncthreads();
  }
  float mu = s1[0] * (1.0f / 768.0f);
  float var = s2[0] * (1.0f / 768.0f) - mu * mu;
  float rstd = rsqrtf(var + 1e-5f);
  bf16_t* o = out + (size_t)row * 768;
  o[t]       = (bf16_t)((v0 - mu) * rstd * gamma[t]       + beta[t]);
  o[t + 256] = (bf16_t)((v1 - mu) * rstd * gamma[t + 256] + beta[t + 256]);
  o[t + 512] = (bf16_t)((v2 - mu) * rstd * gamma[t + 512] + beta[t + 512]);
}

// ---------------------------------------------------------------------------
// QKV projection: (4096x768) @ W^T (768x768). One wave per 32x64 block.
// mode 0/1: out[(b*12+h)*2048+l][e] (BHLE, bf16, *scale)   mode 2: V^T (BHEL)
// ---------------------------------------------------------------------------
__global__ void k_qkv_gemm(const bf16_t* __restrict__ U, const bf16_t* __restrict__ W,
                           bf16_t* __restrict__ out, int mode, float scale) {
  const int m0 = blockIdx.x * 32, n0 = blockIdx.y * 64;
  ACC_INIT(acc);
  gemm_block_2x4(U + (size_t)m0 * 768, 768, W + (size_t)n0 * 768, 768, 768, acc);
  const int lane = threadIdx.x & 31, h = lane >> 4, nn = lane & 15;
#pragma unroll
  for (int i = 0; i < 2; ++i)
#pragma unroll
    for (int r = 0; r < 8; ++r) {
      int mg = m0 + 16 * i + r + 8 * h;
      int bb = mg >> 11, l = mg & 2047;
#pragma unroll
      for (int j = 0; j < 4; ++j) {
        int n = n0 + 16 * j + nn, hh = n >> 6, e = n & 63;
        float v = acc[i][j][r] * scale;
        if (mode == 2)
          out[(((size_t)(bb * 12 + hh) * 64 + e) << 11) + l] = (bf16_t)v;   // V^T
        else
          out[(((size_t)(bb * 12 + hh) * 2048 + l) << 6) + e] = (bf16_t)v;  // Q / K
      }
    }
}

// ---------------------------------------------------------------------------
// Flash attention, tanh-clipped logits. One wave per (b,h, 16-row query tile).
// Q pre-scaled by 1/sqrt(E). V given transposed (BHEL) so B-frags are contiguous.
// ---------------------------------------------------------------------------
__global__ void k_attention(const bf16_t* __restrict__ Q, const bf16_t* __restrict__ K,
                            const bf16_t* __restrict__ VT, const int* __restrict__ amask,
                            bf16_t* __restrict__ heads) {
  const int qt = blockIdx.x;         // query tile 0..127
  const int bh = blockIdx.y;         // 0..23
  const int b = bh / 12, hhead = bh % 12;
  const int lane = threadIdx.x & 31, h = lane >> 4, nn = lane & 15;
  __shared__ bf16_t P[16 * 32];      // probability tile, one wave per block

  const bf16_t* Qt = Q + ((size_t)bh * 2048 + qt * 16) * 64;
  v16bf aq0 = load_Afrag(Qt, 64);        // K(e) = 0..31
  v16bf aq1 = load_Afrag(Qt + 32, 64);   // K(e) = 32..63

  v8f o0 = {}, o1 = {}, o2 = {}, o3 = {};
  float mrow[8], lrow[8];
#pragma unroll
  for (int r = 0; r < 8; ++r) { mrow[r] = -1e30f; lrow[r] = 0.0f; }

  const int nchunks = qt / 2 + 1;    // covers keys 0 .. qt*16+15 (causal)
  for (int kc = 0; kc < nchunks; ++kc) {
    const int key0 = kc * 32;
    const bf16_t* Kt = K + (size_t)bh * 2048 * 64 + (size_t)key0 * 64;
    v8f s0 = {}, s1 = {};
    s0 = wmma_bf16(aq0, load_Bfrag(Kt, 64), s0);                 // keys key0..+15
    s0 = wmma_bf16(aq1, load_Bfrag(Kt + 32, 64), s0);
    s1 = wmma_bf16(aq0, load_Bfrag(Kt + 16 * 64, 64), s1);       // keys key0+16..+31
    s1 = wmma_bf16(aq1, load_Bfrag(Kt + 16 * 64 + 32, 64), s1);

    int mv  = amask[b * 2048 + key0 + lane];
    int mk0 = __shfl(mv, nn, 32);
    int mk1 = __shfl(mv, 16 + nn, 32);

    float t0[8], t1[8];
#pragma unroll
    for (int r = 0; r < 8; ++r) {
      int row = qt * 16 + r + 8 * h;
      int c0 = key0 + nn, c1 = key0 + 16 + nn;
      float a0 = 30.0f * fast_tanh(s0[r]);
      float a1 = 30.0f * fast_tanh(s1[r]);
      t0[r] = (c0 <= row && mk0 != 0) ? a0 : -1e30f;
      t1[r] = (c1 <= row && mk1 != 0) ? a1 : -1e30f;
    }
#pragma unroll
    for (int r = 0; r < 8; ++r) {
      float cm = fmaxf(t0[r], t1[r]);
      cm = fmaxf(cm, __shfl_xor(cm, 1, 32));
      cm = fmaxf(cm, __shfl_xor(cm, 2, 32));
      cm = fmaxf(cm, __shfl_xor(cm, 4, 32));
      cm = fmaxf(cm, __shfl_xor(cm, 8, 32));   // row max within 16-lane half
      float mnew = fmaxf(mrow[r], cm);
      float alpha = __expf(mrow[r] - mnew);
      float p0 = __expf(t0[r] - mnew);
      float p1 = __expf(t1[r] - mnew);
      float ps = p0 + p1;
      ps += __shfl_xor(ps, 1, 32);
      ps += __shfl_xor(ps, 2, 32);
      ps += __shfl_xor(ps, 4, 32);
      ps += __shfl_xor(ps, 8, 32);
      lrow[r] = lrow[r] * alpha + ps;
      mrow[r] = mnew;
      P[(r + 8 * h) * 32 + nn]      = (bf16_t)p0;   // C-layout -> LDS tile
      P[(r + 8 * h) * 32 + 16 + nn] = (bf16_t)p1;
      o0[r] *= alpha; o1[r] *= alpha; o2[r] *= alpha; o3[r] *= alpha;
    }
    // Re-read P in A-fragment layout (LDS ops are in-order within a wave)
    v16bf ap = load_Afrag(P, 32);
    const bf16_t* Vb = VT + (size_t)bh * 64 * 2048 + key0;
    o0 = wmma_bf16(ap, load_Bfrag(Vb + (size_t)0  * 2048, 2048), o0);
    o1 = wmma_bf16(ap, load_Bfrag(Vb + (size_t)16 * 2048, 2048), o1);
    o2 = wmma_bf16(ap, load_Bfrag(Vb + (size_t)32 * 2048, 2048), o2);
    o3 = wmma_bf16(ap, load_Bfrag(Vb + (size_t)48 * 2048, 2048), o3);
  }
#pragma unroll
  for (int r = 0; r < 8; ++r) {
    float inv = 1.0f / lrow[r];
    int l = qt * 16 + r + 8 * h;
    size_t base = ((size_t)(b * 2048 + l)) * 768 + hhead * 64;
    heads[base + 0  + nn] = (bf16_t)(o0[r] * inv);
    heads[base + 16 + nn] = (bf16_t)(o1[r] * inv);
    heads[base + 32 + nn] = (bf16_t)(o2[r] * inv);
    heads[base + 48 + nn] = (bf16_t)(o3[r] * inv);
  }
}

// ---------------------------------------------------------------------------
// W_O GEMM fused with residual: z1 = x + heads @ W_O^T   (f32 out)
// ---------------------------------------------------------------------------
__global__ void k_wo_gemm(const bf16_t* __restrict__ Hd, const bf16_t* __restrict__ Wo,
                          const float* __restrict__ X, float* __restrict__ Z1) {
  const int m0 = blockIdx.x * 32, n0 = blockIdx.y * 64;
  ACC_INIT(acc);
  gemm_block_2x4(Hd + (size_t)m0 * 768, 768, Wo + (size_t)n0 * 768, 768, 768, acc);
  const int lane = threadIdx.x & 31, h = lane >> 4, nn = lane & 15;
#pragma unroll
  for (int i = 0; i < 2; ++i)
#pragma unroll
    for (int r = 0; r < 8; ++r) {
      size_t rowbase = (size_t)(m0 + 16 * i + r + 8 * h) * 768;
#pragma unroll
      for (int j = 0; j < 4; ++j) {
        size_t idx = rowbase + n0 + 16 * j + nn;
        Z1[idx] = X[idx] + acc[i][j][r];
      }
    }
}

// MLP up: v3 = gelu(v1 @ W_up + b_up)  (N=9216, exact GELU, bf16 out)
__global__ void k_up_gemm(const bf16_t* __restrict__ V1, const bf16_t* __restrict__ WupT,
                          const float* __restrict__ bup, bf16_t* __restrict__ V3) {
  const int m0 = blockIdx.x * 32, n0 = blockIdx.y * 64;
  ACC_INIT(acc);
  gemm_block_2x4(V1 + (size_t)m0 * 768, 768, WupT + (size_t)n0 * 768, 768, 768, acc);
  const int lane = threadIdx.x & 31, h = lane >> 4, nn = lane & 15;
#pragma unroll
  for (int i = 0; i < 2; ++i)
#pragma unroll
    for (int r = 0; r < 8; ++r) {
      size_t rowbase = (size_t)(m0 + 16 * i + r + 8 * h) * 9216;
#pragma unroll
      for (int j = 0; j < 4; ++j) {
        int n = n0 + 16 * j + nn;
        float v = acc[i][j][r] + bup[n];
        float g = 0.5f * v * (1.0f + erff(v * 0.70710678118654752f));
        V3[rowbase + n] = (bf16_t)g;
      }
    }
}

// MLP down + residual: out = z1 + v3 @ W_down + b_down   (K=9216, f32 out)
__global__ void k_down_gemm(const bf16_t* __restrict__ V3, const bf16_t* __restrict__ WdT,
                            const float* __restrict__ bdn, const float* __restrict__ Z1,
                            float* __restrict__ out) {
  const int m0 = blockIdx.x * 32, n0 = blockIdx.y * 64;
  ACC_INIT(acc);
  gemm_block_2x4(V3 + (size_t)m0 * 9216, 9216, WdT + (size_t)n0 * 9216, 9216, 9216, acc);
  const int lane = threadIdx.x & 31, h = lane >> 4, nn = lane & 15;
#pragma unroll
  for (int i = 0; i < 2; ++i)
#pragma unroll
    for (int r = 0; r < 8; ++r) {
      size_t rowbase = (size_t)(m0 + 16 * i + r + 8 * h) * 768;
#pragma unroll
      for (int j = 0; j < 4; ++j) {
        int n = n0 + 16 * j + nn;
        size_t idx = rowbase + n;
        out[idx] = Z1[idx] + acc[i][j][r] + bdn[n];
      }
    }
}

// ---------------------------------------------------------------------------
extern "C" void kernel_launch(void* const* d_in, const int* in_sizes, int n_in,
                              void* d_out, int out_size, void* d_ws, size_t ws_size,
                              hipStream_t stream) {
  const float* x    = (const float*)d_in[0];
  const int*   mask = (const int*)d_in[1];
  const float* WQ   = (const float*)d_in[2];
  const float* WK   = (const float*)d_in[3];
  const float* WV   = (const float*)d_in[4];
  const float* g1   = (const float*)d_in[5];
  const float* b1   = (const float*)d_in[6];
  const float* g2   = (const float*)d_in[7];
  const float* b2   = (const float*)d_in[8];
  const float* WO   = (const float*)d_in[9];
  const float* Wup  = (const float*)d_in[10];
  const float* bup  = (const float*)d_in[11];
  const float* Wdn  = (const float*)d_in[12];
  const float* bdn  = (const float*)d_in[13];
  float* out = (float*)d_out;

  char* ws = (char*)d_ws;
  size_t off = 0;
  auto alloc = [&](size_t bytes) -> char* {
    char* p = ws + off;
    off += (bytes + 255) & ~(size_t)255;
    return p;
  };
  const size_t MD = 4096ull * 768;       // rows x D
  bf16_t* u     = (bf16_t*)alloc(MD * 2);
  bf16_t* wqb   = (bf16_t*)alloc(589824ull * 2);
  bf16_t* wkb   = (bf16_t*)alloc(589824ull * 2);
  bf16_t* wvb   = (bf16_t*)alloc(589824ull * 2);
  bf16_t* wob   = (bf16_t*)alloc(589824ull * 2);
  bf16_t* wupT  = (bf16_t*)alloc(768ull * 9216 * 2);
  bf16_t* wdnT  = (bf16_t*)alloc(768ull * 9216 * 2);
  bf16_t* Qb    = (bf16_t*)alloc(MD * 2);   // (B,H,L,E) bf16
  bf16_t* Kb    = (bf16_t*)alloc(MD * 2);   // (B,H,L,E)
  bf16_t* VTb   = (bf16_t*)alloc(MD * 2);   // (B,H,E,L)
  bf16_t* heads = (bf16_t*)alloc(MD * 2);   // (B,L,H*E)
  float*  z1    = (float*)alloc(MD * 4);
  bf16_t* v1b   = (bf16_t*)alloc(MD * 2);
  bf16_t* v3b   = (bf16_t*)alloc(4096ull * 9216 * 2);

  // 1) weight conversions (bf16; W_up/W_down transposed to (N,K)-major)
  k_f32_to_bf16<<<dim3(2304), dim3(256), 0, stream>>>(WQ, wqb, 589824);
  k_f32_to_bf16<<<dim3(2304), dim3(256), 0, stream>>>(WK, wkb, 589824);
  k_f32_to_bf16<<<dim3(2304), dim3(256), 0, stream>>>(WV, wvb, 589824);
  k_f32_to_bf16<<<dim3(2304), dim3(256), 0, stream>>>(WO, wob, 589824);
  k_transpose_bf16<<<dim3(576, 48), dim3(16, 16), 0, stream>>>(Wup, wupT, 768, 9216);
  k_transpose_bf16<<<dim3(48, 576), dim3(16, 16), 0, stream>>>(Wdn, wdnT, 9216, 768);

  // 2) LN1
  k_layernorm<<<dim3(4096), dim3(256), 0, stream>>>(x, g1, b1, u);

  // 3) QKV projections (Q pre-scaled by 1/sqrt(64)); 32x64 blocks per wave
  k_qkv_gemm<<<dim3(128, 12), dim3(32), 0, stream>>>(u, wqb, Qb, 0, 0.125f);
  k_qkv_gemm<<<dim3(128, 12), dim3(32), 0, stream>>>(u, wkb, Kb, 1, 1.0f);
  k_qkv_gemm<<<dim3(128, 12), dim3(32), 0, stream>>>(u, wvb, VTb, 2, 1.0f);

  // 4) attention
  k_attention<<<dim3(128, 24), dim3(32), 0, stream>>>(Qb, Kb, VTb, mask, heads);

  // 5) output projection + residual
  k_wo_gemm<<<dim3(128, 12), dim3(32), 0, stream>>>(heads, wob, x, z1);

  // 6) LN2
  k_layernorm<<<dim3(4096), dim3(256), 0, stream>>>(z1, g2, b2, v1b);

  // 7) MLP
  k_up_gemm<<<dim3(128, 144), dim3(32), 0, stream>>>(v1b, wupT, bup, v3b);
  k_down_gemm<<<dim3(128, 12), dim3(32), 0, stream>>>(v3b, wdnT, bdn, z1, out);
}
